// GoalMlpExtractor_40398462386700
// MI455X (gfx1250) — matrix operand order repaired and checked by reading, not compiled, not run on hardware
//
#include <hip/hip_runtime.h>
#include <stdint.h>

#define G_EXP   16
#define D_DIM   512
#define B_TOK   4096
#define TILE_M  16
#define MAX_TILES (B_TOK / TILE_M + G_EXP)   // 272
#define X_STRIDE 552   // padded bf16 row stride: 1104 B (16B aligned), bank-conflict-free

typedef __attribute__((ext_vector_type(16))) __bf16          v16bf;
typedef __attribute__((ext_vector_type(4)))  __bf16          v4bf;
typedef __attribute__((ext_vector_type(8)))  float           v8f;
typedef __attribute__((ext_vector_type(16))) float           vf16;
typedef __attribute__((ext_vector_type(4)))  float           vf4;
typedef __attribute__((ext_vector_type(8)))  unsigned short  ushort8;

union AFrag {
    v16bf   v;
    ushort8 h[2];
};

// Branch-free tanh: v_exp_f32 is hw exp2 with full range handling.
// exp2 -> +inf => rcp -> 0 => +1 ;  exp2 -> 0 => rcp(1)=1 => -1. Saturates right.
__device__ __forceinline__ float fast_tanh(float x) {
    float t = __builtin_amdgcn_exp2f(x * 2.885390081777927f);  // 2*log2(e)
    return 1.0f - 2.0f * __builtin_amdgcn_rcpf(t + 1.0f);
}

// ---------------------------------------------------------------------------
// Kernel 1: route tokens by goal expert; build expert-contiguous order +
// a list of (expert, start, count<=16) M-tiles in workspace.
// ws layout (ints): [0]=numTiles, [1..272]=tileExpert, [273..544]=tileStart,
//                   [545..816]=tileCnt, [817..817+4096)=order
// ---------------------------------------------------------------------------
__global__ __launch_bounds__(256) void route_kernel(
        const long long* __restrict__ goal, int* __restrict__ ws) {
    __shared__ int cnt[G_EXP];
    __shared__ int cur[G_EXP];
    __shared__ int base[G_EXP];
    const int tid = threadIdx.x;
    if (tid < G_EXP) { cnt[tid] = 0; cur[tid] = 0; }
    __syncthreads();
    for (int t = tid; t < B_TOK; t += blockDim.x) {
        int g = (int)goal[t];
        atomicAdd(&cnt[g], 1);
    }
    __syncthreads();
    if (tid == 0) {
        int* tExp   = ws + 1;
        int* tStart = ws + 1 + MAX_TILES;
        int* tCnt   = ws + 1 + 2 * MAX_TILES;
        int acc = 0, nt = 0;
        for (int g = 0; g < G_EXP; ++g) {
            base[g] = acc;
            int c = cnt[g];
            for (int off = 0; off < c; off += TILE_M) {
                tExp[nt]   = g;
                tStart[nt] = acc + off;
                tCnt[nt]   = (c - off < TILE_M) ? (c - off) : TILE_M;
                ++nt;
            }
            acc += c;
        }
        ws[0] = nt;
    }
    __syncthreads();
    int* order = ws + 1 + 3 * MAX_TILES;
    for (int t = tid; t < B_TOK; t += blockDim.x) {
        int g = (int)goal[t];
        int pos = base[g] + atomicAdd(&cur[g], 1);
        order[pos] = t;
    }
}

// ---------------------------------------------------------------------------
// One Linear(512,512)+tanh layer for a 16-token tile, all 8 waves cooperating.
// A (activations) from LDS bf16 (2x ds_load_b128 per fragment); B (weights)
// gathered from global fp32 columns (L2-resident) and packed-converted to
// bf16 in registers. Wave w owns N-tiles [4w, 4w+4).
// ---------------------------------------------------------------------------
template <bool TO_GLOBAL>
__device__ __forceinline__ void expert_layer(
        const float* __restrict__ W,      // [D_in=512 rows][D_out=512 cols]
        const float* __restrict__ bias,   // [512]
        const unsigned short* __restrict__ Ain,  // LDS bf16 [16][X_STRIDE]
        unsigned short* __restrict__ Aout,       // LDS bf16 (if !TO_GLOBAL)
        float* __restrict__ gout,                // global [token][512] (if TO_GLOBAL)
        const int* __restrict__ tokenIds, int m) {
    const int lane  = threadIdx.x & 31;
    const int wave  = threadIdx.x >> 5;         // 0..7
    const int lhalf = lane >> 4;                // 0 | 1
    const int l15   = lane & 15;
    const int aoff  = lhalf ? 16 : 0;           // A-frag K base offset (contiguous, matches B)
    const int bkoff = lhalf ? 16 : 0;           // B-frag K base offset
    const unsigned short* arowp = Ain + l15 * X_STRIDE;

    #pragma unroll
    for (int nt = 0; nt < 4; ++nt) {
        const int n0   = (wave * 4 + nt) * 16;
        const int ncol = n0 + l15;
        // per-lane base pointer; walked by a constant stride so the 16 element
        // offsets (e*2048 B) fold into the instruction IOFFSET field
        const float* wp = W + ncol + (size_t)bkoff * D_DIM;
        v8f acc = {};
        #pragma unroll 2
        for (int kc = 0; kc < D_DIM / 32; ++kc) {
            const int kbase = kc * 32;
            AFrag a;
            a.h[0] = *(const ushort8*)(arowp + kbase + aoff);        // K 0-7 / 16-23
            a.h[1] = *(const ushort8*)(arowp + kbase + 8 + aoff);    // K 8-15 / 24-31
            vf16 wf;
            #pragma unroll
            for (int e = 0; e < 16; ++e)
                wf[e] = wp[(size_t)e * D_DIM];
            v16bf b = __builtin_convertvector(wf, v16bf);            // v_cvt_pk_bf16_f32
            acc = __builtin_amdgcn_wmma_f32_16x16x32_bf16(
                      false, a.v, false, b, (short)0, acc, false, false);
            wp += (size_t)32 * D_DIM;
        }
        const float bv = bias[ncol];
        #pragma unroll
        for (int v = 0; v < 8; ++v) {
            const float r   = fast_tanh(acc[v] + bv);
            const int   row = v + (lhalf ? 8 : 0);
            if (TO_GLOBAL) {
                if (row < m)
                    gout[(size_t)tokenIds[row] * D_DIM + ncol] = r;
            } else {
                unsigned short rb;
                {
                    __bf16 hb = (__bf16)r;
                    rb = __builtin_bit_cast(unsigned short, hb);
                }
                Aout[row * X_STRIDE + ncol] = rb;
            }
        }
    }
}

// ---------------------------------------------------------------------------
// Kernel 2: one block per 16-token tile of one expert; runs both 2-layer MLPs.
// ---------------------------------------------------------------------------
__global__ __launch_bounds__(256) void mlp_kernel(
        const float* __restrict__ features,
        const float* __restrict__ Wp0, const float* __restrict__ bp0,
        const float* __restrict__ Wp1, const float* __restrict__ bp1,
        const float* __restrict__ Wv0, const float* __restrict__ bv0,
        const float* __restrict__ Wv1, const float* __restrict__ bv1,
        const int* __restrict__ ws,
        float* __restrict__ out_pi, float* __restrict__ out_vf) {
    __shared__ __align__(16) unsigned short Xs[TILE_M * X_STRIDE];
    __shared__ __align__(16) unsigned short Hs[TILE_M * X_STRIDE];
    __shared__ int tokenIds[TILE_M];

    const int tileId = blockIdx.x;
    const int numTiles = ws[0];
    if (tileId >= numTiles) return;

    const int g     = ws[1 + tileId];
    const int start = ws[1 + MAX_TILES + tileId];
    const int m     = ws[1 + 2 * MAX_TILES + tileId];
    const int* order = ws + 1 + 3 * MAX_TILES;

    if (threadIdx.x < TILE_M)
        tokenIds[threadIdx.x] = (threadIdx.x < m) ? order[start + threadIdx.x] : 0;
    __syncthreads();

    // Stage X tile into LDS as bf16: 16 threads per row, 32 floats each,
    // packed-converted and stored as 8-byte groups (ds_store_b64).
    {
        const int row = threadIdx.x >> 4;
        const int seg = (threadIdx.x & 15) * 32;
        unsigned short* dst = &Xs[row * X_STRIDE + seg];
        if (row < m) {
            const vf4* src =
                (const vf4*)(features + (size_t)tokenIds[row] * D_DIM + seg);
            #pragma unroll
            for (int i = 0; i < 8; ++i) {
                v4bf h = __builtin_convertvector(src[i], v4bf);
                *(v4bf*)(dst + i * 4) = h;
            }
        } else {
            #pragma unroll
            for (int i = 0; i < 8; ++i)
                *(v4bf*)(dst + i * 4) = (v4bf){};
        }
    }
    __syncthreads();

    const size_t wstride = (size_t)D_DIM * D_DIM;
    // Policy head
    expert_layer<false>(Wp0 + g * wstride, bp0 + g * D_DIM, Xs, Hs, nullptr, tokenIds, m);
    __syncthreads();
    expert_layer<true >(Wp1 + g * wstride, bp1 + g * D_DIM, Hs, nullptr, out_pi, tokenIds, m);
    __syncthreads();
    // Value head (Xs preserved)
    expert_layer<false>(Wv0 + g * wstride, bv0 + g * D_DIM, Xs, Hs, nullptr, tokenIds, m);
    __syncthreads();
    expert_layer<true >(Wv1 + g * wstride, bv1 + g * D_DIM, Hs, nullptr, out_vf, tokenIds, m);
}

extern "C" void kernel_launch(void* const* d_in, const int* in_sizes, int n_in,
                              void* d_out, int out_size, void* d_ws, size_t ws_size,
                              hipStream_t stream) {
    const float*     features = (const float*)d_in[0];
    const float*     Wp0      = (const float*)d_in[1];
    const float*     bp0      = (const float*)d_in[2];
    const float*     Wp1      = (const float*)d_in[3];
    const float*     bp1      = (const float*)d_in[4];
    const float*     Wv0      = (const float*)d_in[5];
    const float*     bv0      = (const float*)d_in[6];
    const float*     Wv1      = (const float*)d_in[7];
    const float*     bv1      = (const float*)d_in[8];
    const long long* goal     = (const long long*)d_in[9];

    float* out    = (float*)d_out;
    float* out_pi = out;
    float* out_vf = out + (size_t)B_TOK * D_DIM;
    int*   wsI    = (int*)d_ws;

    route_kernel<<<1, 256, 0, stream>>>(goal, wsI);
    mlp_kernel<<<MAX_TILES, 256, 0, stream>>>(features, Wp0, bp0, Wp1, bp1,
                                              Wv0, bv0, Wv1, bv1, wsI,
                                              out_pi, out_vf);
}
